// GINModule_55087250538927
// MI455X (gfx1250) — matrix operand order repaired and checked by reading, not compiled, not run on hardware
//
#include <hip/hip_runtime.h>

#define N_NODES  100000
#define N_EDGES  400000
#define N_GRAPHS 2000
#define F0       84
#define F2       840
#define FH       1024
#define FOUT     384

// padded K (multiple of 32) per GEMM
#define KP_GIN   96      // K=84   -> 3 ksteps
#define KP_FC1   1696    // K=1680 -> 53 ksteps
#define KP_FC2   1024    // K=1024 -> 32 ksteps

typedef __attribute__((ext_vector_type(16))) __bf16 v16bf;
typedef __attribute__((ext_vector_type(8)))  __bf16 v8bf;
typedef __attribute__((ext_vector_type(8)))  float  v8f;

// float -> bf16, round-to-nearest-even
static __device__ __forceinline__ __bf16 f2bf(float f) {
    union { float f; unsigned u; } x; x.f = f;
    unsigned r = x.u + 0x7fffu + ((x.u >> 16) & 1u);
    unsigned short s = (unsigned short)(r >> 16);
    union { unsigned short s; __bf16 b; } y; y.s = s;
    return y.b;
}

__global__ void fill_f32(float* __restrict__ p, float v, int n) {
    int i = blockIdx.x * blockDim.x + threadIdx.x;
    if (i < n) p[i] = v;
}

// agg[dst[e], f] += h[src[e], f]
__global__ void scatter_add_kernel(const float* __restrict__ h,
                                   const int* __restrict__ src,
                                   const int* __restrict__ dst,
                                   float* __restrict__ agg,
                                   int nE, int F) {
    int i = blockIdx.x * blockDim.x + threadIdx.x;
    int total = nE * F;
    if (i >= total) return;
    int e = i / F;
    int f = i - e * F;
    atomicAdd(&agg[(size_t)dst[e] * F + f], h[(size_t)src[e] * F + f]);
}

static __device__ __forceinline__ void atomicMaxF(float* addr, float val) {
    unsigned* ua = (unsigned*)addr;
    unsigned old = *ua;
    while (__uint_as_float(old) < val) {
        unsigned assumed = old;
        old = atomicCAS(ua, assumed, __float_as_uint(val));
        if (old == assumed) break;
    }
}

// segment max + sum pooling of h2 [N, F2] over batch -> pmax/psum [G, F2]
__global__ void pool_kernel(const float* __restrict__ h2,
                            const int* __restrict__ batch,
                            float* __restrict__ pmax,
                            float* __restrict__ psum) {
    int i = blockIdx.x * blockDim.x + threadIdx.x;
    const int total = N_NODES * F2;
    if (i >= total) return;
    int n = i / F2;
    int f = i - n * F2;
    int g = batch[n];
    float v = h2[(size_t)n * F2 + f];
    atomicAdd(&psum[(size_t)g * F2 + f], v);
    atomicMaxF(&pmax[(size_t)g * F2 + f], v);
}

// Stage A = (a1 + a2) as bf16, row-major, K padded F -> KP (zeros)
__global__ void stage_add_bf16(const float* __restrict__ a1,
                               const float* __restrict__ a2,
                               __bf16* __restrict__ out, int M, int F, int KP) {
    int i = blockIdx.x * blockDim.x + threadIdx.x;
    int total = M * KP;
    if (i >= total) return;
    int row = i / KP;
    int k   = i - row * KP;
    float v = 0.0f;
    if (k < F) v = a1[(size_t)row * F + k] + a2[(size_t)row * F + k];
    out[i] = f2bf(v);
}

// Stage A = concat(a1, a2) as bf16, K padded 2F -> KP (zeros)
__global__ void stage_concat_bf16(const float* __restrict__ a1,
                                  const float* __restrict__ a2,
                                  __bf16* __restrict__ out, int M, int F, int KP) {
    int i = blockIdx.x * blockDim.x + threadIdx.x;
    int total = M * KP;
    if (i >= total) return;
    int row = i / KP;
    int k   = i - row * KP;
    float v = 0.0f;
    if (k < F)          v = a1[(size_t)row * F + k];
    else if (k < 2 * F) v = a2[(size_t)row * F + (k - F)];
    out[i] = f2bf(v);
}

// Pack weight W[K,Ncols] (row-major) into per-fragment bf16 layout:
// Bp[((ntile*ksteps + ks)*32 + lane)*16 + e] = W[ks*32 + (e>>3)*16 + (lane>>4)*8 + (e&7), ntile*16 + (lane&15)]
// zero-padded outside K x Ncols (ntile count may be padded). ISA 7.12.2 16-bit B layout.
__global__ void pack_b_kernel(const float* __restrict__ W, __bf16* __restrict__ Bp,
                              int K, int Ncols, int ksteps, int total) {
    int i = blockIdx.x * blockDim.x + threadIdx.x;
    if (i >= total) return;
    int e    = i & 15;
    int lane = (i >> 4) & 31;
    int rem  = i >> 9;            // nt*ksteps + ks
    int ks   = rem % ksteps;
    int nt   = rem / ksteps;
    int col  = nt * 16 + (lane & 15);
    int k    = ks * 32 + (e >> 3) * 16 + (lane >> 4) * 8 + (e & 7);
    float v = 0.0f;
    if (k < K && col < Ncols) v = W[(size_t)k * Ncols + col];
    Bp[i] = f2bf(v);
}

// WMMA GEMM, register-blocked 1x4 tiles per wave, A tile staged in LDS via
// async global->LDS copies (ASYNCcnt path, ISA 15.18.3):
// C[M,Ncols] = act( A[M, 32*KSTEPS] @ B + bias )
// A: staged bf16, row-major, lda = 32*KSTEPS (padded). M multiple of 16.
// Bp: pre-packed bf16 fragments, tile count padded to multiple of 4.
template <int KSTEPS, bool RELU, bool OUTBF16>
__global__ __launch_bounds__(128)
void wmma_gemm4(const __bf16* __restrict__ A,
                const __bf16* __restrict__ Bp,
                const float* __restrict__ bias,
                float* __restrict__ Cf, __bf16* __restrict__ Cbf, int ldc,
                int Ncols, int nGroups) {
    constexpr int LDA = 32 * KSTEPS;
    constexpr int P   = LDA + 8;   // LDS row pitch (bf16): +16B pad -> conflict-free b128 reads

    __shared__ __attribute__((aligned(16))) __bf16 atile[16 * P];

    const int lane  = threadIdx.x & 31;
    const int wave  = threadIdx.x >> 5;
    const int mtile = blockIdx.x;
    const int grp   = blockIdx.y * 4 + wave;

    // ---- cooperative async copy of the 16 x LDA A tile into LDS (all waves) ----
    {
        const __bf16* abase = A + (size_t)mtile * 16 * LDA;
        const int chunks = 4 * KSTEPS * 16;            // 16-byte chunks in tile
        for (int c = threadIdx.x; c < chunks; c += 128) {
            int rw = c / (4 * KSTEPS);
            int ci = c - rw * (4 * KSTEPS);
            unsigned lds = (unsigned)(unsigned long long)&atile[rw * P + ci * 8];
            unsigned long long ga =
                (unsigned long long)(const void*)(abase + (size_t)rw * LDA + ci * 8);
            asm volatile("global_load_async_to_lds_b128 %0, %1, off"
                         :: "v"(lds), "v"(ga) : "memory");
        }
        asm volatile("s_wait_asynccnt 0x0" ::: "memory");
        __syncthreads();
    }

    if (grp < nGroups) {
        const int nb   = grp * 4;                  // base ntile of this wave's 4 tiles
        const int r    = lane & 15;
        const int half = lane >> 4;                // 0: lanes 0-15, 1: lanes 16-31

        const __bf16* lp = &atile[r * P + half * 8];
        const __bf16* bp = Bp + ((size_t)nb * KSTEPS * 32 + lane) * 16;

        __builtin_prefetch(bp, 0, 1);              // global_prefetch_b8 of B tiles

        v8f acc[4] = {};
#pragma unroll
        for (int ks = 0; ks < KSTEPS; ++ks) {
            v16bf a;
            *((v8bf*)&a)     = *(const v8bf*)(lp + ks * 32);       // K [kb,    kb+8)
            *((v8bf*)&a + 1) = *(const v8bf*)(lp + ks * 32 + 16);  // K [kb+16, kb+24)
#pragma unroll
            for (int j = 0; j < 4; ++j) {
                v16bf b = *(const v16bf*)(bp + (size_t)(j * KSTEPS + ks) * 512);
                acc[j] = __builtin_amdgcn_wmma_f32_16x16x32_bf16(
                    /*neg_a=*/false, a, /*neg_b=*/false, b,
                    /*c_mod=*/(short)0, acc[j], /*reuse_a=*/false, /*reuse_b=*/false);
            }
        }

        // Epilogue: bias + optional ReLU; C layout: col = lane&15, row = vgpr (+8 hi half)
#pragma unroll
        for (int j = 0; j < 4; ++j) {
            int col = (nb + j) * 16 + r;
            float bb = 0.0f;
            if (bias && col < Ncols) bb = bias[col];
#pragma unroll
            for (int v = 0; v < 8; ++v) {
                int rr = mtile * 16 + v + half * 8;
                float val = acc[j][v] + bb;
                if (RELU) val = fmaxf(val, 0.0f);
                if (OUTBF16) {
                    if (col < Ncols) Cbf[(size_t)rr * ldc + col] = f2bf(val);
                } else if (col < Ncols) {
                    Cf[(size_t)rr * ldc + col] = val;
                }
            }
        }
    }
}

static inline size_t alignF(size_t nFloats) { return (nFloats + 15) & ~(size_t)15; } // 64B align

extern "C" void kernel_launch(void* const* d_in, const int* in_sizes, int n_in,
                              void* d_out, int out_size, void* d_ws, size_t ws_size,
                              hipStream_t stream) {
    const float* x     = (const float*)d_in[0];
    const int*   ei    = (const int*)  d_in[1];
    const int*   batch = (const int*)  d_in[2];
    const float* W1    = (const float*)d_in[3];
    const float* b1    = (const float*)d_in[4];
    const float* W2    = (const float*)d_in[5];
    const float* b2    = (const float*)d_in[6];
    const float* Wf1   = (const float*)d_in[7];
    const float* bf1   = (const float*)d_in[8];
    const float* Wf2   = (const float*)d_in[9];
    const float* bf2   = (const float*)d_in[10];

    const int* src = ei;            // edge_index[0]
    const int* dst = ei + N_EDGES;  // edge_index[1]

    float* out_xg = (float*)d_out;                                // [G, 384]
    float* out_h2 = (float*)d_out + (size_t)N_GRAPHS * FOUT;      // [N, 840]

    // tile counts, padded to multiples of 4 for 1x4 register blocking
    const int TN1p  = 8;    // ceil(84/16)=6  -> 8
    const int TN2p  = 56;   // ceil(840/16)=53 -> 56
    const int TNF1  = 64;   // 1024/16
    const int TNF2  = 24;   // 384/16

    // workspace layout (float units, 64B-aligned blocks)
    float* ws = (float*)d_ws;
    size_t off = 0;
    float* agg  = ws + off; off += alignF((size_t)N_NODES * F0);
    float* h1   = ws + off; off += alignF((size_t)N_NODES * F0);
    float* pmax = ws + off; off += alignF((size_t)N_GRAPHS * F2);
    float* psum = ws + off; off += alignF((size_t)N_GRAPHS * F2);
    // bf16 buffers (2 bf16 per float slot)
    size_t aStageMax = (size_t)N_NODES * KP_GIN;                  // >= G*KP_FC1
    __bf16* aStage = (__bf16*)(ws + off); off += alignF((aStageMax + 1) / 2);
    __bf16* hidBf  = (__bf16*)(ws + off); off += alignF(((size_t)N_GRAPHS * FH + 1) / 2);
    const size_t szW1p  = (size_t)TN1p * 3  * 512;
    const size_t szW2p  = (size_t)TN2p * 3  * 512;
    const size_t szWf1p = (size_t)TNF1 * 53 * 512;
    const size_t szWf2p = (size_t)TNF2 * 32 * 512;
    __bf16* W1p  = (__bf16*)(ws + off); off += alignF((szW1p  + 1) / 2);
    __bf16* W2p  = (__bf16*)(ws + off); off += alignF((szW2p  + 1) / 2);
    __bf16* Wf1p = (__bf16*)(ws + off); off += alignF((szWf1p + 1) / 2);
    __bf16* Wf2p = (__bf16*)(ws + off); off += alignF((szWf2p + 1) / 2);

    const int MT_N = N_NODES / 16;    // 6250 (exact)
    const int MT_G = N_GRAPHS / 16;   // 125  (exact)

    // ---- pack all weights into fragment layout (tiny, L2-resident) ----
    pack_b_kernel<<<((int)szW1p  + 255) / 256, 256, 0, stream>>>(W1,  W1p,  F0,     F0,   3,  (int)szW1p);
    pack_b_kernel<<<((int)szW2p  + 255) / 256, 256, 0, stream>>>(W2,  W2p,  F0,     F2,   3,  (int)szW2p);
    pack_b_kernel<<<((int)szWf1p + 255) / 256, 256, 0, stream>>>(Wf1, Wf1p, 2 * F2, FH,   53, (int)szWf1p);
    pack_b_kernel<<<((int)szWf2p + 255) / 256, 256, 0, stream>>>(Wf2, Wf2p, FH,     FOUT, 32, (int)szWf2p);

    // ---- conv1: agg = scatter_add(x); h1 = relu((x+agg)@W1 + b1) ----
    { int n = N_NODES * F0;
      fill_f32<<<(n + 255) / 256, 256, 0, stream>>>(agg, 0.0f, n); }
    { int n = N_EDGES * F0;
      scatter_add_kernel<<<(n + 255) / 256, 256, 0, stream>>>(x, src, dst, agg, N_EDGES, F0); }
    { int n = N_NODES * KP_GIN;
      stage_add_bf16<<<(n + 255) / 256, 256, 0, stream>>>(x, agg, aStage, N_NODES, F0, KP_GIN); }
    { const int ng = TN1p / 4;  // 2
      dim3 g(MT_N, (ng + 3) / 4);
      wmma_gemm4<3, true, false><<<g, 128, 0, stream>>>(aStage, W1p, b1, h1, nullptr,
                                                        F0, F0, ng); }

    // ---- conv2: agg = scatter_add(h1); h2 = relu((h1+agg)@W2 + b2) -> d_out ----
    { int n = N_NODES * F0;
      fill_f32<<<(n + 255) / 256, 256, 0, stream>>>(agg, 0.0f, n); }
    { int n = N_EDGES * F0;
      scatter_add_kernel<<<(n + 255) / 256, 256, 0, stream>>>(h1, src, dst, agg, N_EDGES, F0); }
    { int n = N_NODES * KP_GIN;
      stage_add_bf16<<<(n + 255) / 256, 256, 0, stream>>>(h1, agg, aStage, N_NODES, F0, KP_GIN); }
    { const int ng = TN2p / 4;  // 14
      dim3 g(MT_N, (ng + 3) / 4);
      wmma_gemm4<3, true, false><<<g, 128, 0, stream>>>(aStage, W2p, b2, out_h2, nullptr,
                                                        F2, F2, ng); }

    // ---- pooling: segment max / sum over batch ----
    { int n = N_GRAPHS * F2;
      fill_f32<<<(n + 255) / 256, 256, 0, stream>>>(pmax, -__builtin_inff(), n);
      fill_f32<<<(n + 255) / 256, 256, 0, stream>>>(psum, 0.0f, n); }
    { int n = N_NODES * F2;
      pool_kernel<<<(n + 255) / 256, 256, 0, stream>>>(out_h2, batch, pmax, psum); }

    // ---- fc_g: hid = relu(concat(pmax,psum)@Wf1 + bf1) (bf16 out); xg = hid@Wf2 + bf2 ----
    { int n = N_GRAPHS * KP_FC1;
      stage_concat_bf16<<<(n + 255) / 256, 256, 0, stream>>>(pmax, psum, aStage,
                                                             N_GRAPHS, F2, KP_FC1); }
    { const int ng = TNF1 / 4;  // 16
      dim3 g(MT_G, (ng + 3) / 4);
      wmma_gemm4<53, true, true><<<g, 128, 0, stream>>>(aStage, Wf1p, bf1, nullptr, hidBf,
                                                        FH, FH, ng); }
    { const int ng = TNF2 / 4;  // 6
      dim3 g(MT_G, (ng + 3) / 4);
      wmma_gemm4<32, false, false><<<g, 128, 0, stream>>>(hidBf, Wf2p, bf2, out_xg, nullptr,
                                                          FOUT, FOUT, ng); }
}
